// P20TinyControlConditioner_47064251630310
// MI455X (gfx1250) — compile-verified, matrix operand order checked by reading, not run on hardware
//
#include <hip/hip_runtime.h>
#include <hip/hip_bf16.h>

// ---- problem dims (static) ----
#define B_      4
#define S_      4096
#define DM      2048
#define HEADS_  16
#define BW_     256
#define PAIRS   128          // BW/2
#define ROWS    (B_*S_)      // 16384
#define NOUT    (3*DM + HEADS_)  // 6160
#define NC      64           // number of scan chunks
#define CL      64           // chunk length (NC*CL == S_)

static const size_t QSZ = (size_t)ROWS * DM;   // elements per q/k/v region

// ---- workspace layout (bytes) ----
static const size_t OFF_PROJ  = 0;                                   // f32 [ROWS][BW]
static const size_t OFF_YBF   = OFF_PROJ + (size_t)ROWS*BW_*4;       // bf16 [ROWS][BW]
static const size_t OFF_WINBF = OFF_YBF  + (size_t)ROWS*BW_*2;       // bf16 [BW][DM]
static const size_t OFF_WDBF  = OFF_WINBF + (size_t)BW_*DM*2;        // bf16 [NOUT][BW]
static const size_t OFF_RAMP  = OFF_WDBF + (size_t)NOUT*BW_*2;       // f32 [NOUT]
static const size_t OFF_CARRY = OFF_RAMP + (size_t)NOUT*4;           // f32 [B*P][NC][2]
static const size_t OFF_PREF  = OFF_CARRY + (size_t)B_*PAIRS*NC*2*4; // f32 [B*P][NC][2]

typedef __attribute__((ext_vector_type(16))) __bf16        v16bf;
typedef __attribute__((ext_vector_type(2)))  __bf16        v2bf;
typedef __attribute__((ext_vector_type(8)))  unsigned int  v8ui;
typedef __attribute__((ext_vector_type(8)))  float         v8f;

// native bf16 conversion: pack two f32 -> u32 of two bf16 (backend should
// select v_cvt_pk_bf16_f32 since gfx1250 has native BF16 ALU)
__device__ __forceinline__ unsigned int pack_bf2(float x, float y) {
    v2bf v;
    v[0] = (__bf16)x;
    v[1] = (__bf16)y;
    return __builtin_bit_cast(unsigned int, v);
}

// ---------------- helper conversion kernels ----------------
// n_pairs = n/2 elements processed as f32 pairs -> one u32 of 2 bf16
__global__ void k_f32_to_bf16(const float* __restrict__ src,
                              unsigned int* __restrict__ dst, int n_pairs) {
    int i = blockIdx.x * blockDim.x + threadIdx.x;
    if (i < n_pairs) dst[i] = pack_bf2(src[2*i], src[2*i + 1]);
}

__global__ void k_build_ramp(const float* __restrict__ qkv_logit,  // [3*DM] flat
                             const float* __restrict__ beta_logit, // [HEADS]
                             float* __restrict__ ramp) {
    int i = blockIdx.x * blockDim.x + threadIdx.x;
    if (i >= NOUT) return;
    float lg = (i < 3*DM) ? qkv_logit[i] : beta_logit[i - 3*DM];
    ramp[i] = 1.0f / (1.0f + expf(-lg));
}

// ---------------- GEMM1: proj = hidden x W_in^T (bf16 WMMA) ----------------
// per wave: one 16-row M tile x 64 cols (4 N-tiles). waves = 1024 * 4.
__global__ void k_gemm1(const float* __restrict__ hidden,
                        const unsigned short* __restrict__ winbf,
                        float* __restrict__ proj) {
    const int lane = threadIdx.x & 31;
    const int gw   = blockIdx.x * (blockDim.x >> 5) + (threadIdx.x >> 5);
    const int mt   = gw >> 2;          // 0..1023
    const int ng   = gw & 3;           // 0..3  (64-col group)
    const int lm   = lane & 15;
    const int half = lane >> 4;

    const int row = mt * 16 + lm;
    const float* __restrict__ arow = hidden + (size_t)row * DM;

    v8f acc[4] = {};
    for (int ks = 0; ks < DM / 32; ++ks) {
        const int kb = ks * 32;
        v8ui au;
        #pragma unroll
        for (int v = 0; v < 8; ++v) {
            int k0 = kb + ((v >> 2) << 4) + (half << 3) + ((v & 3) << 1);
            au[v] = pack_bf2(arow[k0], arow[k0 + 1]);
        }
        v16bf a = __builtin_bit_cast(v16bf, au);
        #pragma unroll
        for (int nt = 0; nt < 4; ++nt) {
            const int col = ng * 64 + nt * 16 + lm;
            const unsigned int* __restrict__ bp =
                (const unsigned int*)(winbf + (size_t)col * DM);
            v8ui bu;
            #pragma unroll
            for (int v = 0; v < 8; ++v) {
                int k0 = kb + ((v >> 2) << 4) + (half << 3) + ((v & 3) << 1);
                bu[v] = bp[k0 >> 1];
            }
            v16bf b = __builtin_bit_cast(v16bf, bu);
            acc[nt] = __builtin_amdgcn_wmma_f32_16x16x32_bf16(
                false, a, false, b, (short)0, acc[nt], false, false);
        }
    }
    #pragma unroll
    for (int nt = 0; nt < 4; ++nt) {
        const int col = ng * 64 + nt * 16 + lm;
        #pragma unroll
        for (int r = 0; r < 8; ++r) {
            int grow = mt * 16 + half * 8 + r;
            proj[(size_t)grow * BW_ + col] = acc[nt][r];
        }
    }
}

// ---------------- scan phase A: per-chunk local carries ----------------
// block = 128 threads (one per pair), grid = B*NC; coalesced row reads.
__global__ void k_carries(const float* __restrict__ proj,
                          const float* __restrict__ theta,
                          float* __restrict__ carries) {
    const int b = blockIdx.x >> 6;
    const int c = blockIdx.x & 63;
    const int p = threadIdx.x;
    const float th = theta[p];
    const float cs = cosf(th), sn = sinf(th);
    float s0 = 0.f, s1 = 0.f;
    const float* __restrict__ base =
        proj + ((size_t)(b * S_ + c * CL)) * BW_ + 2 * p;
    for (int t = 0; t < CL; ++t) {
        float x0 = base[0], x1 = base[1];
        float n0 = cs * s0 - sn * s1 + x0;
        float n1 = sn * s0 + cs * s1 + x1;
        s0 = n0; s1 = n1;
        base += BW_;
    }
    size_t idx = ((size_t)(b * PAIRS + p) * NC + c) * 2;
    carries[idx] = s0; carries[idx + 1] = s1;
}

// ---------------- scan phase B: sequential chunk-prefix combine ----------------
__global__ void k_prefix(const float* __restrict__ carries,
                         const float* __restrict__ theta,
                         float* __restrict__ prefixes) {
    const int bp = blockIdx.x * blockDim.x + threadIdx.x;   // 0..511
    const int p  = bp & (PAIRS - 1);
    const float a64 = 64.0f * theta[p];                     // R^CL = rot by CL*theta
    const float c64 = cosf(a64), s64 = sinf(a64);
    float p0 = 0.f, p1 = 0.f;
    for (int c = 0; c < NC; ++c) {
        size_t idx = ((size_t)bp * NC + c) * 2;
        prefixes[idx] = p0; prefixes[idx + 1] = p1;
        float r0 = c64 * p0 - s64 * p1;
        float r1 = s64 * p0 + c64 * p1;
        p0 = r0 + carries[idx];
        p1 = r1 + carries[idx + 1];
    }
}

// ---------------- scan phase C: replay + fused RMSNorm -> bf16 y ----------------
__global__ void k_scan_norm(const float* __restrict__ proj,
                            const float* __restrict__ theta,
                            const float* __restrict__ prefixes,
                            const float* __restrict__ rmsw,
                            unsigned int* __restrict__ ybf_u32) {
    const int b = blockIdx.x >> 6;
    const int c = blockIdx.x & 63;
    const int p = threadIdx.x;                   // 0..127
    const float th = theta[p];
    const float cs = cosf(th), sn = sinf(th);
    const float w0 = rmsw[2 * p], w1 = rmsw[2 * p + 1];
    size_t pidx = ((size_t)(b * PAIRS + p) * NC + c) * 2;
    float s0 = prefixes[pidx], s1 = prefixes[pidx + 1];

    __shared__ float part[4];
    const int wv = threadIdx.x >> 5;
    const int ln = threadIdx.x & 31;
    int row = b * S_ + c * CL;
    const float* __restrict__ base = proj + (size_t)row * BW_ + 2 * p;

    for (int t = 0; t < CL; ++t) {
        float x0 = base[0], x1 = base[1];
        float n0 = cs * s0 - sn * s1 + x0;
        float n1 = sn * s0 + cs * s1 + x1;
        s0 = n0; s1 = n1;
        // block-wide sum of squares (256 values across 128 lanes)
        float v = s0 * s0 + s1 * s1;
        #pragma unroll
        for (int off = 16; off > 0; off >>= 1) v += __shfl_xor(v, off, 32);
        if (ln == 0) part[wv] = v;
        __syncthreads();
        float sum = part[0] + part[1] + part[2] + part[3];
        __syncthreads();
        float rms = rsqrtf(sum * (1.0f / (float)BW_) + 1e-6f);
        ybf_u32[(size_t)(row + t) * (BW_ / 2) + p] =
            pack_bf2(s0 * rms * w0, s1 * rms * w1);
        base += BW_;
    }
}

// ---------------- GEMM2: d = y x W_delta^T, ramp epilogue, split write ----------------
// per wave: 16 rows x 80 cols (5 N-tiles; 385 N-tiles total = 77 groups exactly)
__global__ void k_gemm2(const unsigned short* __restrict__ ybf,
                        const unsigned short* __restrict__ wdbf,
                        const float* __restrict__ ramp,
                        float* __restrict__ dout) {
    const int lane = threadIdx.x & 31;
    const int gw   = blockIdx.x * (blockDim.x >> 5) + (threadIdx.x >> 5);
    const int mt   = gw / 77;          // 0..1023
    const int ng   = gw % 77;          // 0..76
    const int lm   = lane & 15;
    const int half = lane >> 4;

    const int row = mt * 16 + lm;
    const unsigned int* __restrict__ yrow =
        (const unsigned int*)ybf + (size_t)row * (BW_ / 2);

    v8f acc[5] = {};
    for (int ks = 0; ks < BW_ / 32; ++ks) {
        const int kb = ks * 32;
        v8ui au;
        #pragma unroll
        for (int v = 0; v < 8; ++v) {
            int k0 = kb + ((v >> 2) << 4) + (half << 3) + ((v & 3) << 1);
            au[v] = yrow[k0 >> 1];
        }
        v16bf a = __builtin_bit_cast(v16bf, au);
        #pragma unroll
        for (int nt = 0; nt < 5; ++nt) {
            const int col = (ng * 5 + nt) * 16 + lm;
            const unsigned int* __restrict__ bp =
                (const unsigned int*)(wdbf + (size_t)col * BW_);
            v8ui bu;
            #pragma unroll
            for (int v = 0; v < 8; ++v) {
                int k0 = kb + ((v >> 2) << 4) + (half << 3) + ((v & 3) << 1);
                bu[v] = bp[k0 >> 1];
            }
            v16bf b = __builtin_bit_cast(v16bf, bu);
            acc[nt] = __builtin_amdgcn_wmma_f32_16x16x32_bf16(
                false, a, false, b, (short)0, acc[nt], false, false);
        }
    }
    // epilogue: ramp-scale + route tile to q/k/v/beta regions (tiles never straddle)
    #pragma unroll
    for (int nt = 0; nt < 5; ++nt) {
        const int col = (ng * 5 + nt) * 16 + lm;
        const float scale = ramp[col];
        size_t roff; int rstride, lcol;
        if (col < 3 * DM) { roff = (size_t)(col / DM) * QSZ; rstride = DM; lcol = col % DM; }
        else              { roff = 3 * QSZ; rstride = HEADS_; lcol = col - 3 * DM; }
        #pragma unroll
        for (int r = 0; r < 8; ++r) {
            int grow = mt * 16 + half * 8 + r;
            dout[roff + (size_t)grow * rstride + lcol] = acc[nt][r] * scale;
        }
    }
}

// ---------------- host launch ----------------
extern "C" void kernel_launch(void* const* d_in, const int* in_sizes, int n_in,
                              void* d_out, int out_size, void* d_ws, size_t ws_size,
                              hipStream_t stream) {
    const float* hidden     = (const float*)d_in[0];
    const float* W_in       = (const float*)d_in[1];
    const float* theta      = (const float*)d_in[2];
    const float* rmsw       = (const float*)d_in[3];
    const float* W_delta    = (const float*)d_in[4];
    const float* qkv_logit  = (const float*)d_in[5];
    const float* beta_logit = (const float*)d_in[6];
    float* out = (float*)d_out;
    char*  ws  = (char*)d_ws;

    float*          proj    = (float*)(ws + OFF_PROJ);
    unsigned short* ybf     = (unsigned short*)(ws + OFF_YBF);
    unsigned short* winbf   = (unsigned short*)(ws + OFF_WINBF);
    unsigned short* wdbf    = (unsigned short*)(ws + OFF_WDBF);
    float*          ramp    = (float*)(ws + OFF_RAMP);
    float*          carries = (float*)(ws + OFF_CARRY);
    float*          prefix  = (float*)(ws + OFF_PREF);

    // weight conversions (L2-resident afterwards) + ramp precompute
    k_f32_to_bf16<<<((BW_*DM/2) + 255)/256, 256, 0, stream>>>(
        W_in, (unsigned int*)winbf, BW_*DM/2);
    k_f32_to_bf16<<<((NOUT*BW_/2) + 255)/256, 256, 0, stream>>>(
        W_delta, (unsigned int*)wdbf, NOUT*BW_/2);
    k_build_ramp<<<(NOUT + 255)/256, 256, 0, stream>>>(qkv_logit, beta_logit, ramp);

    // GEMM1: 1024 M-tiles x 4 groups = 4096 waves, 8 waves/block
    k_gemm1<<<4096/8, 256, 0, stream>>>(hidden, winbf, proj);

    // rotary scan: carries -> prefix combine -> fused replay + RMSNorm
    k_carries<<<B_*NC, PAIRS, 0, stream>>>(proj, theta, carries);
    k_prefix<<<(B_*PAIRS)/128, 128, 0, stream>>>(carries, theta, prefix);
    k_scan_norm<<<B_*NC, PAIRS, 0, stream>>>(proj, theta, prefix, rmsw,
                                             (unsigned int*)ybf);

    // GEMM2: 1024 M-tiles x 77 groups = 78848 waves, 8 waves/block
    k_gemm2<<<78848/8, 256, 0, stream>>>(ybf, wdbf, ramp, out);
}
